// Attention_34144990003509
// MI455X (gfx1250) — matrix operand order repaired and checked by reading, not compiled, not run on hardware
//
#include <hip/hip_runtime.h>

typedef _Float16 half_t;
typedef __attribute__((ext_vector_type(16))) _Float16 v16h;
typedef __attribute__((ext_vector_type(8)))  _Float16 v8h;
typedef __attribute__((ext_vector_type(4)))  _Float16 v4h;
typedef __attribute__((ext_vector_type(8)))  float    v8f;
typedef __attribute__((ext_vector_type(4)))  float    v4f;

#define B_  4
#define N_  1024
#define D_  1024
#define H_  16
#define DH_ 64

static __device__ __forceinline__ v16h cat16(v8h lo, v8h hi) {
  v16h r;
#pragma unroll
  for (int i = 0; i < 8; ++i) { r[i] = lo[i]; r[i + 8] = hi[i]; }
  return r;
}

static __device__ __forceinline__ v8h ldg8(const half_t* p) {
  return *(const v8h*)p;
}

// ---------------- Kernel 1: fp32 -> fp16 conversion (x4 vectorized) ----------
__global__ void cvt_f32_f16_x4(const float* __restrict__ s, half_t* __restrict__ d, int n4) {
  int i = blockIdx.x * blockDim.x + threadIdx.x;
  if (i < n4) {
    v4f x = ((const v4f*)s)[i];
    v4h y;
#pragma unroll
    for (int j = 0; j < 4; ++j) y[j] = (half_t)x[j];
    ((v4h*)d)[i] = y;
  }
}

// ---------------- Kernel 2: QKV projection GEMM (32x64 per wave) -------------
// qkv[m][e] = sum_k x[m][k] * w_qkv[e][k];  M=4096, K=1024, Nout=3072
// Q,K stored [B,H,N,dh]; V stored transposed [B,H,dh,N].
__global__ void __launch_bounds__(128) qkv_gemm(const half_t* __restrict__ xh,
                                                const half_t* __restrict__ wh,
                                                half_t* __restrict__ Qh,
                                                half_t* __restrict__ Kh,
                                                half_t* __restrict__ Vt) {
  const int lane  = threadIdx.x & 31;
  const int wave  = __builtin_amdgcn_readfirstlane(threadIdx.x >> 5);
  const int tile  = blockIdx.x * 4 + wave;        // scalar
  const int NT    = (3 * D_) / 64;                // 48
  const int mt    = tile / NT;
  const int nt    = tile % NT;
  const int m0    = mt * 32;
  const int e0    = nt * 64;                      // 64-aligned: same s & head for whole tile
  const int lrow  = lane & 15;
  const int lhalf = lane >> 4;

  const half_t* a0 = xh + (m0 + lrow) * D_ + lhalf * 16;      // rows m0..m0+15
  const half_t* a1 = a0 + 16 * D_;                            // rows m0+16..m0+31
  const half_t* bp0 = wh + (e0 +  0 + lrow) * D_ + lhalf * 16;
  const half_t* bp1 = wh + (e0 + 16 + lrow) * D_ + lhalf * 16;
  const half_t* bp2 = wh + (e0 + 32 + lrow) * D_ + lhalf * 16;
  const half_t* bp3 = wh + (e0 + 48 + lrow) * D_ + lhalf * 16;

  v8f acc[2][4] = {};
#pragma unroll 2
  for (int kb = 0; kb < D_; kb += 32) {
    v16h A0 = cat16(ldg8(a0 + kb), ldg8(a0 + kb + 8));
    v16h A1 = cat16(ldg8(a1 + kb), ldg8(a1 + kb + 8));
    v16h B0 = cat16(ldg8(bp0 + kb), ldg8(bp0 + kb + 8));
    v16h B1 = cat16(ldg8(bp1 + kb), ldg8(bp1 + kb + 8));
    v16h B2 = cat16(ldg8(bp2 + kb), ldg8(bp2 + kb + 8));
    v16h B3 = cat16(ldg8(bp3 + kb), ldg8(bp3 + kb + 8));
    acc[0][0] = __builtin_amdgcn_wmma_f32_16x16x32_f16(false, A0, false, B0, (short)0, acc[0][0], false, false);
    acc[0][1] = __builtin_amdgcn_wmma_f32_16x16x32_f16(false, A0, false, B1, (short)0, acc[0][1], false, false);
    acc[0][2] = __builtin_amdgcn_wmma_f32_16x16x32_f16(false, A0, false, B2, (short)0, acc[0][2], false, false);
    acc[0][3] = __builtin_amdgcn_wmma_f32_16x16x32_f16(false, A0, false, B3, (short)0, acc[0][3], false, false);
    acc[1][0] = __builtin_amdgcn_wmma_f32_16x16x32_f16(false, A1, false, B0, (short)0, acc[1][0], false, false);
    acc[1][1] = __builtin_amdgcn_wmma_f32_16x16x32_f16(false, A1, false, B1, (short)0, acc[1][1], false, false);
    acc[1][2] = __builtin_amdgcn_wmma_f32_16x16x32_f16(false, A1, false, B2, (short)0, acc[1][2], false, false);
    acc[1][3] = __builtin_amdgcn_wmma_f32_16x16x32_f16(false, A1, false, B3, (short)0, acc[1][3], false, false);
  }

  // Whole tile maps to one of Q/K/V, one batch block, one head (all scalar).
  const int su = e0 >> 10;                 // 0=Q 1=K 2=V
  const int hu = (e0 & 1023) >> 6;         // head
  const int bu = m0 >> 10;                 // batch
  const int bh = bu * H_ + hu;

  if (su == 2) {
#pragma unroll
    for (int i = 0; i < 2; ++i)
#pragma unroll
      for (int r = 0; r < 8; ++r) {
        const int n = (m0 + i * 16 + r + lhalf * 8) & 1023;
#pragma unroll
        for (int j = 0; j < 4; ++j) {
          const int dd = j * 16 + lrow;
          Vt[(bh * DH_ + dd) * N_ + n] = (half_t)acc[i][j][r];
        }
      }
  } else {
    half_t* dst = (su == 0) ? Qh : Kh;
#pragma unroll
    for (int i = 0; i < 2; ++i)
#pragma unroll
      for (int r = 0; r < 8; ++r) {
        const int n = (m0 + i * 16 + r + lhalf * 8) & 1023;
#pragma unroll
        for (int j = 0; j < 4; ++j) {
          const int dd = j * 16 + lrow;
          dst[(bh * N_ + n) * DH_ + dd] = (half_t)acc[i][j][r];
        }
      }
  }
}

// ---------------- Kernel 3: flash attention ------------------------
// One wave handles a 16-query tile for one (b,h); loops keys in chunks of 32.
__global__ void __launch_bounds__(128) flash_attn(const half_t* __restrict__ Qh,
                                                  const half_t* __restrict__ Kh,
                                                  const half_t* __restrict__ Vt,
                                                  half_t* __restrict__ AOh) {
  __shared__ __align__(16) half_t pl[4][16 * 32];   // per-wave P tile (row-major 16x32)
  const int lane  = threadIdx.x & 31;
  const int wave  = __builtin_amdgcn_readfirstlane(threadIdx.x >> 5);
  const int tid   = blockIdx.x * 4 + wave;          // scalar
  const int b     = tid >> 10;                      // / (H_ * 64)
  const int rest  = tid & 1023;
  const int h     = rest >> 6;
  const int qt    = rest & 63;
  const int q0    = qt * 16;
  const int lrow  = lane & 15;
  const int lhalf = lane >> 4;

  const half_t* Qp = Qh + (size_t)(b * H_ + h) * N_ * DH_;
  const half_t* Kp = Kh + (size_t)(b * H_ + h) * N_ * DH_;
  const half_t* Vp = Vt + (size_t)(b * H_ + h) * DH_ * N_;
  half_t* pw = pl[wave];

  // Q A-operands for the two 32-wide d chunks
  v16h aq0, aq1;
  {
    const half_t* qa = Qp + (q0 + lrow) * DH_ + lhalf * 16;
    aq0 = cat16(ldg8(qa),      ldg8(qa + 8));
    aq1 = cat16(ldg8(qa + 32), ldg8(qa + 40));
  }

  float mrun[8], lrun[8];
  v8f oacc[4] = {};
#pragma unroll
  for (int r = 0; r < 8; ++r) { mrun[r] = -1e30f; lrun[r] = 0.0f; }

  for (int key0 = 0; key0 < N_; key0 += 32) {
    // scores: two 16x16 tiles (key cols 0-15 and 16-31 of this chunk)
    v8f s0 = {}, s1 = {};
    {
      const half_t* kb0 = Kp + (key0 + lrow) * DH_ + lhalf * 16;
      v16h b0 = cat16(ldg8(kb0),      ldg8(kb0 + 8));
      v16h b1 = cat16(ldg8(kb0 + 32), ldg8(kb0 + 40));
      s0 = __builtin_amdgcn_wmma_f32_16x16x32_f16(false, aq0, false, b0, (short)0, s0, false, false);
      s0 = __builtin_amdgcn_wmma_f32_16x16x32_f16(false, aq1, false, b1, (short)0, s0, false, false);
    }
    {
      const half_t* kb1 = Kp + (key0 + 16 + lrow) * DH_ + lhalf * 16;
      v16h b0 = cat16(ldg8(kb1),      ldg8(kb1 + 8));
      v16h b1 = cat16(ldg8(kb1 + 32), ldg8(kb1 + 40));
      s1 = __builtin_amdgcn_wmma_f32_16x16x32_f16(false, aq0, false, b0, (short)0, s1, false, false);
      s1 = __builtin_amdgcn_wmma_f32_16x16x32_f16(false, aq1, false, b1, (short)0, s1, false, false);
    }

    const float scale = 0.125f;   // d_head^-0.5
    float p0[8], p1[8], corr[8];
#pragma unroll
    for (int r = 0; r < 8; ++r) {
      float a  = s0[r] * scale;
      float bq = s1[r] * scale;
      float mx = fmaxf(a, bq);
      mx = fmaxf(mx, __shfl_xor(mx, 1));
      mx = fmaxf(mx, __shfl_xor(mx, 2));
      mx = fmaxf(mx, __shfl_xor(mx, 4));
      mx = fmaxf(mx, __shfl_xor(mx, 8));
      float mnew = fmaxf(mrun[r], mx);
      corr[r] = __expf(mrun[r] - mnew);
      p0[r]   = __expf(a  - mnew);
      p1[r]   = __expf(bq - mnew);
      float rs = p0[r] + p1[r];
      rs += __shfl_xor(rs, 1);
      rs += __shfl_xor(rs, 2);
      rs += __shfl_xor(rs, 4);
      rs += __shfl_xor(rs, 8);
      lrun[r] = lrun[r] * corr[r] + rs;
      mrun[r] = mnew;
    }
#pragma unroll
    for (int t = 0; t < 4; ++t)
#pragma unroll
      for (int r = 0; r < 8; ++r) oacc[t][r] *= corr[r];

    // C-layout -> A-layout repack of P through LDS (row-major [16][32] f16)
#pragma unroll
    for (int r = 0; r < 8; ++r) {
      const int row = r + lhalf * 8;
      pw[row * 32 + lrow]      = (half_t)p0[r];
      pw[row * 32 + lrow + 16] = (half_t)p1[r];
    }
    asm volatile("s_wait_dscnt 0x0" ::: "memory");
    v16h pa;
    {
      const half_t* pp = pw + lrow * 32 + lhalf * 16;
      pa = cat16(*(const v8h*)pp, *(const v8h*)(pp + 8));
    }

    // PV: accumulate 16x64 output, 4 d-column tiles
#pragma unroll
    for (int t = 0; t < 4; ++t) {
      const half_t* vb = Vp + (t * 16 + lrow) * N_ + key0 + lhalf * 16;
      v16h bv = cat16(ldg8(vb), ldg8(vb + 8));
      oacc[t] = __builtin_amdgcn_wmma_f32_16x16x32_f16(false, pa, false, bv, (short)0, oacc[t], false, false);
    }
  }

  // finalize: divide by row sums, store attention output as [B,N,D] f16
  half_t* aop = AOh + (size_t)b * N_ * D_;
#pragma unroll
  for (int t = 0; t < 4; ++t) {
#pragma unroll
    for (int r = 0; r < 8; ++r) {
      const int n   = q0 + r + lhalf * 8;
      const int col = h * DH_ + t * 16 + lrow;
      aop[n * D_ + col] = (half_t)(oacc[t][r] / lrun[r]);
    }
  }
}

// ---------------- Kernel 4: output projection + bias (32x64 per wave) --------
__global__ void __launch_bounds__(128) out_gemm(const half_t* __restrict__ AOh,
                                                const half_t* __restrict__ wh,
                                                const float* __restrict__ bias,
                                                float* __restrict__ out) {
  const int lane  = threadIdx.x & 31;
  const int wave  = __builtin_amdgcn_readfirstlane(threadIdx.x >> 5);
  const int tile  = blockIdx.x * 4 + wave;        // scalar
  const int NT    = D_ / 64;                      // 16
  const int mt    = tile / NT;
  const int nt    = tile % NT;
  const int m0    = mt * 32;
  const int e0    = nt * 64;
  const int lrow  = lane & 15;
  const int lhalf = lane >> 4;

  const half_t* a0 = AOh + (m0 + lrow) * D_ + lhalf * 16;
  const half_t* a1 = a0 + 16 * D_;
  const half_t* bp0 = wh + (e0 +  0 + lrow) * D_ + lhalf * 16;
  const half_t* bp1 = wh + (e0 + 16 + lrow) * D_ + lhalf * 16;
  const half_t* bp2 = wh + (e0 + 32 + lrow) * D_ + lhalf * 16;
  const half_t* bp3 = wh + (e0 + 48 + lrow) * D_ + lhalf * 16;

  v8f acc[2][4] = {};
#pragma unroll 2
  for (int kb = 0; kb < D_; kb += 32) {
    v16h A0 = cat16(ldg8(a0 + kb), ldg8(a0 + kb + 8));
    v16h A1 = cat16(ldg8(a1 + kb), ldg8(a1 + kb + 8));
    v16h B0 = cat16(ldg8(bp0 + kb), ldg8(bp0 + kb + 8));
    v16h B1 = cat16(ldg8(bp1 + kb), ldg8(bp1 + kb + 8));
    v16h B2 = cat16(ldg8(bp2 + kb), ldg8(bp2 + kb + 8));
    v16h B3 = cat16(ldg8(bp3 + kb), ldg8(bp3 + kb + 8));
    acc[0][0] = __builtin_amdgcn_wmma_f32_16x16x32_f16(false, A0, false, B0, (short)0, acc[0][0], false, false);
    acc[0][1] = __builtin_amdgcn_wmma_f32_16x16x32_f16(false, A0, false, B1, (short)0, acc[0][1], false, false);
    acc[0][2] = __builtin_amdgcn_wmma_f32_16x16x32_f16(false, A0, false, B2, (short)0, acc[0][2], false, false);
    acc[0][3] = __builtin_amdgcn_wmma_f32_16x16x32_f16(false, A0, false, B3, (short)0, acc[0][3], false, false);
    acc[1][0] = __builtin_amdgcn_wmma_f32_16x16x32_f16(false, A1, false, B0, (short)0, acc[1][0], false, false);
    acc[1][1] = __builtin_amdgcn_wmma_f32_16x16x32_f16(false, A1, false, B1, (short)0, acc[1][1], false, false);
    acc[1][2] = __builtin_amdgcn_wmma_f32_16x16x32_f16(false, A1, false, B2, (short)0, acc[1][2], false, false);
    acc[1][3] = __builtin_amdgcn_wmma_f32_16x16x32_f16(false, A1, false, B3, (short)0, acc[1][3], false, false);
  }

#pragma unroll
  for (int i = 0; i < 2; ++i)
#pragma unroll
    for (int r = 0; r < 8; ++r) {
      const int row = m0 + i * 16 + r + lhalf * 8;
#pragma unroll
      for (int j = 0; j < 4; ++j) {
        const int col = e0 + j * 16 + lrow;
        out[row * D_ + col] = acc[i][j][r] + bias[col];
      }
    }
}

extern "C" void kernel_launch(void* const* d_in, const int* in_sizes, int n_in,
                              void* d_out, int out_size, void* d_ws, size_t ws_size,
                              hipStream_t stream) {
  (void)in_sizes; (void)n_in; (void)out_size; (void)ws_size;
  const float* x     = (const float*)d_in[0];
  const float* w_qkv = (const float*)d_in[1];
  const float* w_out = (const float*)d_in[2];
  const float* b_out = (const float*)d_in[3];
  float* out = (float*)d_out;

  half_t* ws    = (half_t*)d_ws;
  half_t* xh    = ws;                        // 4M halves (reused as attention output)
  half_t* wqkvh = ws + 4 * 1024 * 1024;      // 3M
  half_t* wouth = ws + 7 * 1024 * 1024;      // 1M
  half_t* Qh    = ws + 8 * 1024 * 1024;      // 4M
  half_t* Kh    = ws + 12 * 1024 * 1024;     // 4M
  half_t* Vt    = ws + 16 * 1024 * 1024;     // 4M (transposed [B,H,dh,N])

  cvt_f32_f16_x4<<<(1024 * 1024) / 256, 256, 0, stream>>>(x,     xh,    1024 * 1024);
  cvt_f32_f16_x4<<<(768  * 1024) / 256, 256, 0, stream>>>(w_qkv, wqkvh, 768 * 1024);
  cvt_f32_f16_x4<<<(256  * 1024) / 256, 256, 0, stream>>>(w_out, wouth, 256 * 1024);

  // 4096x3072 GEMM: (4096/32)*(3072/64) = 128*48 = 6144 tiles, 4 waves/block
  qkv_gemm<<<1536, 128, 0, stream>>>(xh, wqkvh, Qh, Kh, Vt);
  // 4*16*64 = 4096 q-tiles, 4 waves/block
  flash_attn<<<1024, 128, 0, stream>>>(Qh, Kh, Vt, xh);
  // 4096x1024 GEMM: 128*16 = 2048 tiles
  out_gemm<<<512, 128, 0, stream>>>(xh, wouth, b_out, out);
}
